// NonlinearStringTensorForce_86947317941187
// MI455X (gfx1250) — matrix operand order repaired and checked by reading, not compiled, not run on hardware
//
#include <hip/hip_runtime.h>

// CDNA5 / gfx1250, wave32.
typedef __attribute__((ext_vector_type(16))) __bf16 v16bf;
typedef __attribute__((ext_vector_type(8)))  float  v8f;

#define NMODES 64
#define MM2    4096   // 64*64: GEMM M (=m*64+i) and K (=j*64+k) extents
#define BATCH  1024

__device__ __forceinline__ unsigned short f32_to_bf16_rne(float f) {
    unsigned int u = __float_as_uint(f);
    unsigned int r = u + 0x7FFFu + ((u >> 16) & 1u);   // round-to-nearest-even
    return (unsigned short)(r >> 16);
}

union BFrag {
    v16bf  v;
    uint4  u[2];
};

union Pack8 {
    unsigned short s[8];
    uint4 v;
};

// ---------------------------------------------------------------------------
// Kernel 0: Abf = bf16(A), row-major [4096][4096]. One-time convert so the
// GEMM hot loop is pure b128 loads + WMMA (A is re-read 16x from L2 as bf16).
// ---------------------------------------------------------------------------
__global__ __launch_bounds__(256)
void convert_a_kernel(const float* __restrict__ A, unsigned short* __restrict__ Abf) {
    size_t base = ((size_t)blockIdx.x * 256 + threadIdx.x) * 8;
    const float4 f0 = *reinterpret_cast<const float4*>(A + base);
    const float4 f1 = *reinterpret_cast<const float4*>(A + base + 4);
    Pack8 p;
    p.s[0] = f32_to_bf16_rne(f0.x); p.s[1] = f32_to_bf16_rne(f0.y);
    p.s[2] = f32_to_bf16_rne(f0.z); p.s[3] = f32_to_bf16_rne(f0.w);
    p.s[4] = f32_to_bf16_rne(f1.x); p.s[5] = f32_to_bf16_rne(f1.y);
    p.s[6] = f32_to_bf16_rne(f1.z); p.s[7] = f32_to_bf16_rne(f1.w);
    *reinterpret_cast<uint4*>(Abf + base) = p.v;
}

// ---------------------------------------------------------------------------
// Kernel 1: P[b][jk] = bf16( q[b, jk>>6] * q[b, jk&63] ), layout [1024][4096].
// ---------------------------------------------------------------------------
__global__ __launch_bounds__(256)
void build_p_kernel(const float* __restrict__ q, unsigned short* __restrict__ P) {
    int base = (blockIdx.x * 256 + threadIdx.x) * 4;
    int b  = base >> 12;
    int jk = base & 4095;
    int j  = jk >> 6;
    int k  = jk & 63;             // multiple of 4; +3 stays within one j block
    float qj = q[b * NMODES + j];
    const float4 qk = *reinterpret_cast<const float4*>(q + b * NMODES + k);
    ushort4 r;
    r.x = f32_to_bf16_rne(qj * qk.x);
    r.y = f32_to_bf16_rne(qj * qk.y);
    r.z = f32_to_bf16_rne(qj * qk.z);
    r.w = f32_to_bf16_rne(qj * qk.w);
    *reinterpret_cast<ushort4*>(P + base) = r;
}

// ---------------------------------------------------------------------------
// Kernel 2: out[b, m] = - sum_i q[b,i] * ( sum_jk Abf[m*64+i, jk] * P[b, jk] )
// Workgroup = 2 m-values (rows) x 64 batch columns. 4 waves; each wave owns
// one 16-row tile per m (2 A fragments) x 4 col-tiles (B fragments shared
// across both m's) => 8 WMMAs per 32-wide K step. Register double-buffered.
// ---------------------------------------------------------------------------
__global__ __launch_bounds__(128)
void tensor_force_kernel(const unsigned short* __restrict__ Abf,
                         const float* __restrict__ q,
                         const unsigned short* __restrict__ P,
                         float* __restrict__ out) {
    const int batchBase = blockIdx.x * 64;   // 16 batch groups
    const int m0        = blockIdx.y * 2;    // 32 m-pairs

    const int tid  = threadIdx.x;
    const int wave = tid >> 5;               // row tile i in [wave*16, wave*16+16)
    const int lane = tid & 31;
    const int half = lane >> 4;              // ISA 16-bit A/B lane split
    const int l16  = lane & 15;              // A: row in tile; B/C/D: column

    // A fragment rows (bf16): per-lane K runs at [half*8, +8) and [16+half*8, +8).
    const unsigned short* Arow[2];
    Arow[0] = Abf + (size_t)(m0 * NMODES + wave * 16 + l16) * MM2 + half * 8;
    Arow[1] = Arow[0] + (size_t)NMODES * MM2;          // m0 + 1

    // B fragment rows: lane column b, K runs at [half*8, +8) and [16+half*8, +8).
    const unsigned short* Prow[4];
#pragma unroll
    for (int t = 0; t < 4; ++t)
        Prow[t] = P + (size_t)(batchBase + t * 16 + l16) * MM2 + half * 8;

    v8f acc[2][4];
#pragma unroll
    for (int s = 0; s < 2; ++s)
#pragma unroll
        for (int t = 0; t < 4; ++t)
#pragma unroll
            for (int r = 0; r < 8; ++r) acc[s][t][r] = 0.0f;

    // ---- prologue: fragments for kb = 0
    BFrag aC[2], bC[4];
#pragma unroll
    for (int s = 0; s < 2; ++s) {
        aC[s].u[0] = *reinterpret_cast<const uint4*>(Arow[s]);
        aC[s].u[1] = *reinterpret_cast<const uint4*>(Arow[s] + 16);
    }
#pragma unroll
    for (int t = 0; t < 4; ++t) {
        bC[t].u[0] = *reinterpret_cast<const uint4*>(Prow[t]);
        bC[t].u[1] = *reinterpret_cast<const uint4*>(Prow[t] + 16);
    }

#pragma unroll 2
    for (int kb = 0; kb < MM2; kb += 32) {
        int kn = kb + 32;
        if (kn == MM2) kn = 0;               // tail reload (harmless, uniform)

        // ---- issue next iteration's 12x b128 loads first (double buffer)
        BFrag aN[2], bN[4];
#pragma unroll
        for (int s = 0; s < 2; ++s) {
            aN[s].u[0] = *reinterpret_cast<const uint4*>(Arow[s] + kn);
            aN[s].u[1] = *reinterpret_cast<const uint4*>(Arow[s] + kn + 16);
        }
#pragma unroll
        for (int t = 0; t < 4; ++t) {
            bN[t].u[0] = *reinterpret_cast<const uint4*>(Prow[t] + kn);
            bN[t].u[1] = *reinterpret_cast<const uint4*>(Prow[t] + kn + 16);
        }

        // ---- 8 WMMAs on current fragments (B reused across both m's)
#pragma unroll
        for (int s = 0; s < 2; ++s)
#pragma unroll
            for (int t = 0; t < 4; ++t)
                acc[s][t] = __builtin_amdgcn_wmma_f32_16x16x32_bf16(
                    false, aC[s].v, false, bC[t].v, (short)0, acc[s][t],
                    false, false);

#pragma unroll
        for (int s = 0; s < 2; ++s) aC[s] = aN[s];
#pragma unroll
        for (int t = 0; t < 4; ++t) bC[t] = bN[t];
    }

    // ---- Epilogue: out[b, m] = -sum_i D[i, b] * q[b, i]
    // C/D layout: VGPR r holds row M = r + half*8 in the wave's 16-row tile,
    // column = l16. Global i = wave*16 + half*8 + r, r = 0..7 contiguous.
    __shared__ float red[2][4][4][32];       // [m][colTile][wave][lane], 4 KB
#pragma unroll
    for (int t = 0; t < 4; ++t) {
        const int b = batchBase + t * 16 + l16;
        const float* qw = q + (size_t)b * NMODES + wave * 16 + half * 8;
        const float4 w0 = *reinterpret_cast<const float4*>(qw);
        const float4 w1 = *reinterpret_cast<const float4*>(qw + 4);
#pragma unroll
        for (int s = 0; s < 2; ++s) {
            float p = acc[s][t][0] * w0.x + acc[s][t][1] * w0.y +
                      acc[s][t][2] * w0.z + acc[s][t][3] * w0.w +
                      acc[s][t][4] * w1.x + acc[s][t][5] * w1.y +
                      acc[s][t][6] * w1.z + acc[s][t][7] * w1.w;
            red[s][t][wave][lane] = p;
        }
    }
    __syncthreads();

    // 128 outputs per workgroup: one thread each.
    {
        const int s  = tid >> 6;             // which m
        const int c2 = tid & 63;             // column within batch group
        const int t  = c2 >> 4;
        const int c  = c2 & 15;
        float sum = 0.0f;
#pragma unroll
        for (int w = 0; w < 4; ++w)
            sum += red[s][t][w][c] + red[s][t][w][c + 16];
        out[(size_t)(batchBase + c2) * NMODES + (m0 + s)] = -sum;
    }
}

// ---------------------------------------------------------------------------
extern "C" void kernel_launch(void* const* d_in, const int* in_sizes, int n_in,
                              void* d_out, int out_size, void* d_ws, size_t ws_size,
                              hipStream_t stream) {
    const float* q = (const float*)d_in[0];          // [1024, 64] f32
    const float* A = (const float*)d_in[1];          // [64^4] f32 (64 MB)
    float* out = (float*)d_out;                      // [1024, 64] f32

    unsigned short* Abf = (unsigned short*)d_ws;                       // 32 MB
    unsigned short* P   = (unsigned short*)((char*)d_ws +
                           (size_t)MM2 * MM2 * sizeof(unsigned short)); // +8 MB

    // Kernel 0: A f32 -> bf16 (16.7M elems, 8/thread)
    convert_a_kernel<<<dim3((MM2 * MM2) / (256 * 8)), dim3(256), 0, stream>>>(A, Abf);

    // Kernel 1: P[b][jk] = bf16(q[b,j] * q[b,k]) (4.2M elems, 4/thread)
    build_p_kernel<<<dim3((BATCH * MM2) / (256 * 4)), dim3(256), 0, stream>>>(q, P);

    // Kernel 2: WMMA GEMM + fused final contraction
    tensor_force_kernel<<<dim3(BATCH / 64, NMODES / 2), dim3(128), 0, stream>>>(Abf, q, P, out);
}
